// Codebook1_33535104647886
// MI455X (gfx1250) — matrix operand order repaired and checked by reading, not compiled, not run on hardware
//
#include <hip/hip_runtime.h>

// ---------------------------------------------------------------------------
// VQ codebook forward for MI455X (gfx1250, wave32, WMMA).
// d = 2 - 2*zn.en^T since rows are unit-normalized  =>  argmin d == argmax zn@en^T
// loss = (1+beta) * mean((zq - zl)^2);  out = emb[idx] @ Wp + bp.
// All GEMMs via v_wmma_f32_16x16x32_bf16 (f32 accumulate).
// ---------------------------------------------------------------------------

typedef __attribute__((ext_vector_type(16))) __bf16        v16bf;
typedef __attribute__((ext_vector_type(8)))  float         v8f;
typedef __attribute__((ext_vector_type(8)))  unsigned int  v8u;

#define N_ROWS   16384   // B*R
#define IN_D     512
#define LAT_D    256
#define OUT_D    512
#define N_CODES  8192

__device__ __forceinline__ v8f wmma_bf16(v16bf a, v16bf b, v8f c) {
  return __builtin_amdgcn_wmma_f32_16x16x32_bf16(
      /*neg_a=*/false, a, /*neg_b=*/false, b,
      /*c_mod=*/(short)0, c, /*reuse_a=*/false, /*reuse_b=*/false);
}

// A operand: 16x32 bf16 tile, rows of a row-major matrix.
// ISA layout: lane = 16*h + m; VGPR r holds K pair at 16*(r>>2) + 8*h + 2*(r&3).
__device__ __forceinline__ v16bf load_a_bf16(const __bf16* base, int ld,
                                             int row0, int k0, int lane) {
  int h = (lane >> 4) & 1, m = lane & 15;
  const unsigned int* rp =
      (const unsigned int*)(base + (size_t)(row0 + m) * ld);
  v8u u;
#pragma unroll
  for (int r = 0; r < 8; ++r) {
    int kb = k0 + ((r >> 2) << 4) + (h << 3) + ((r & 3) << 1);
    u[r] = rp[kb >> 1];
  }
  return __builtin_bit_cast(v16bf, u);
}

// A operand loaded from f32 source with on-the-fly bf16 conversion.
__device__ __forceinline__ v16bf load_a_f32(const float* base, int ld,
                                            int row0, int k0, int lane) {
  int h = (lane >> 4) & 1, m = lane & 15;
  const float* rp = base + (size_t)(row0 + m) * ld;
  v16bf a;
#pragma unroll
  for (int r = 0; r < 8; ++r) {
    int kb = k0 + ((r >> 2) << 4) + (h << 3) + ((r & 3) << 1);
    float2 f = *(const float2*)(rp + kb);
    a[2 * r]     = (__bf16)f.x;
    a[2 * r + 1] = (__bf16)f.y;
  }
  return a;
}

// B operand: 32x16 bf16 tile (K x N) where B[k][n] = M2[n][k], M2 row-major.
// Layout: lane = 16*h + n; VGPR r holds K pair at 16*(r>>2) + 8*h + 2*(r&3).
__device__ __forceinline__ v16bf load_b_bf16(const __bf16* base, int ld,
                                             int n0, int k0, int lane) {
  int h = (lane >> 4) & 1, n = lane & 15;
  const unsigned int* rp =
      (const unsigned int*)(base + (size_t)(n0 + n) * ld);
  v8u u;
#pragma unroll
  for (int r = 0; r < 8; ++r) {
    int kb = k0 + ((r >> 2) << 4) + (h << 3) + ((r & 3) << 1);
    u[r] = rp[kb >> 1];
  }
  return __builtin_bit_cast(v16bf, u);
}

// Running argmax update for one 16x16 score tile held in C layout.
__device__ __forceinline__ void upd_best(const v8f& c, int code,
                                         float* best, int* bidx) {
#pragma unroll
  for (int j = 0; j < 8; ++j) {
    float v = c[j];
    if (v > best[j] || (v == best[j] && code < bidx[j])) {
      best[j] = v; bidx[j] = code;
    }
  }
}

// ---------------------------------------------------------------------------
// Transpose + f32->bf16: src is K x N row-major, dst is N x K row-major.
__global__ __launch_bounds__(256) void transpose_cvt(
    const float* __restrict__ src, __bf16* __restrict__ dst, int K, int N) {
  int i = blockIdx.x * 256 + threadIdx.x;
  if (i < K * N) {
    int k = i / N, n = i - k * N;
    dst[(size_t)n * K + k] = (__bf16)src[i];
  }
}

// Row L2-normalize (256 cols, f32 in) -> bf16 out. One wave per row.
__global__ __launch_bounds__(256) void row_l2norm_bf16(
    const float* __restrict__ src, __bf16* __restrict__ dst) {
  int lane = threadIdx.x & 31, wave = threadIdx.x >> 5;
  int row = blockIdx.x * 8 + wave;
  const float* rp = src + (size_t)row * LAT_D + lane * 8;
  float4 x0 = *(const float4*)(rp);
  float4 x1 = *(const float4*)(rp + 4);
  float s = x0.x * x0.x + x0.y * x0.y + x0.z * x0.z + x0.w * x0.w +
            x1.x * x1.x + x1.y * x1.y + x1.z * x1.z + x1.w * x1.w;
#pragma unroll
  for (int m = 16; m >= 1; m >>= 1) s += __shfl_xor(s, m, 32);
  float scale = 1.0f / fmaxf(sqrtf(s), 1e-12f);
  __bf16* op = dst + (size_t)row * LAT_D + lane * 8;
  op[0] = (__bf16)(x0.x * scale); op[1] = (__bf16)(x0.y * scale);
  op[2] = (__bf16)(x0.z * scale); op[3] = (__bf16)(x0.w * scale);
  op[4] = (__bf16)(x1.x * scale); op[5] = (__bf16)(x1.y * scale);
  op[6] = (__bf16)(x1.z * scale); op[7] = (__bf16)(x1.w * scale);
}

// ---------------------------------------------------------------------------
// GEMM1: zl = z @ Wq + bq.  Per wave: 16 rows x 64 cols, K=512.
// 8 waves/block: 2 M-tiles x 4 col-chunks. grid = N_ROWS/32.
__global__ __launch_bounds__(256) void g1_gemm(
    const float* __restrict__ z, const __bf16* __restrict__ WqT,
    const float* __restrict__ bq, float* __restrict__ zl) {
  int lane = threadIdx.x & 31, wave = threadIdx.x >> 5;
  int row0 = (blockIdx.x * 2 + (wave >> 2)) * 16;
  int col0 = (wave & 3) * 64;
  v8f acc[4] = {};
  for (int k0 = 0; k0 < IN_D; k0 += 32) {
    // Issue all operand loads before the WMMA chain so waits stagger.
    v16bf B[4];
#pragma unroll
    for (int t = 0; t < 4; ++t)
      B[t] = load_b_bf16(WqT, IN_D, col0 + t * 16, k0, lane);
    v16bf a = load_a_f32(z, IN_D, row0, k0, lane);
#pragma unroll
    for (int t = 0; t < 4; ++t) acc[t] = wmma_bf16(a, B[t], acc[t]);
  }
  int h = lane >> 4, n = lane & 15;
#pragma unroll
  for (int t = 0; t < 4; ++t) {
    int col = col0 + t * 16 + n;
    float bias = bq[col];
#pragma unroll
    for (int j = 0; j < 8; ++j)
      zl[(size_t)(row0 + 8 * h + j) * LAT_D + col] = acc[t][j] + bias;
  }
}

// ---------------------------------------------------------------------------
// Distance + argmax: scores = zn @ en^T, argmax over 8192 codes (== argmin d).
// Block = 8 waves covering 64 rows: wave = (M-tile 0..3) x (code-half 0..1).
// Waves sharing a code-half issue identical B addresses -> WGP$ reuse, so L2
// traffic for en drops to ~1 GB total. A strip (16x256) stays in 64 VGPRs;
// B is double-buffered so the 8-WMMA chain overlaps the next tile's loads.
// grid = N_ROWS/64.
__global__ __launch_bounds__(256) void g2_argmax(
    const __bf16* __restrict__ zn, const __bf16* __restrict__ en,
    int* __restrict__ idx_i, float* __restrict__ idx_f) {
  __shared__ float s_best[64][2];
  __shared__ int   s_bidx[64][2];
  int lane = threadIdx.x & 31, wave = threadIdx.x >> 5;
  int mt = wave >> 1;          // 0..3  : 16-row tile within block
  int half = wave & 1;         // 0..1  : code range half
  int row0 = blockIdx.x * 64 + mt * 16;
  int code0 = half * (N_CODES / 2);
  const int NT = (N_CODES / 2) / 16;   // 256 code tiles per wave

  v16bf A[8];
#pragma unroll
  for (int ks = 0; ks < 8; ++ks)
    A[ks] = load_a_bf16(zn, LAT_D, row0, ks * 32, lane);

  float best[8]; int bidx[8];
#pragma unroll
  for (int j = 0; j < 8; ++j) { best[j] = -3.0e38f; bidx[j] = 0x7fffffff; }

  v16bf B0[8];
#pragma unroll
  for (int ks = 0; ks < 8; ++ks)
    B0[ks] = load_b_bf16(en, LAT_D, code0, ks * 32, lane);

  for (int t = 0; t < NT - 1; ++t) {
    v16bf B1[8];
#pragma unroll
    for (int ks = 0; ks < 8; ++ks)
      B1[ks] = load_b_bf16(en, LAT_D, code0 + (t + 1) * 16, ks * 32, lane);
    v8f c = {};
#pragma unroll
    for (int ks = 0; ks < 8; ++ks) c = wmma_bf16(A[ks], B0[ks], c);
    upd_best(c, code0 + t * 16 + (lane & 15), best, bidx);
#pragma unroll
    for (int ks = 0; ks < 8; ++ks) B0[ks] = B1[ks];
  }
  {  // final tile (no prefetch)
    v8f c = {};
#pragma unroll
    for (int ks = 0; ks < 8; ++ks) c = wmma_bf16(A[ks], B0[ks], c);
    upd_best(c, code0 + (NT - 1) * 16 + (lane & 15), best, bidx);
  }

  // reduce across the 16 lanes that share rows 8h..8h+7
#pragma unroll
  for (int m = 8; m >= 1; m >>= 1) {
#pragma unroll
    for (int j = 0; j < 8; ++j) {
      float ov = __shfl_xor(best[j], m, 32);
      int   oi = __shfl_xor(bidx[j], m, 32);
      if (ov > best[j] || (ov == best[j] && oi < bidx[j])) { best[j] = ov; bidx[j] = oi; }
    }
  }
  int h = lane >> 4;
  if ((lane & 15) == 0) {
#pragma unroll
    for (int j = 0; j < 8; ++j) {
      s_best[mt * 16 + 8 * h + j][half] = best[j];
      s_bidx[mt * 16 + 8 * h + j][half] = bidx[j];
    }
  }
  __syncthreads();
  if (threadIdx.x < 64) {
    int r = threadIdx.x;
    float bv = s_best[r][0]; int bi = s_bidx[r][0];
    float v1 = s_best[r][1]; int i1 = s_bidx[r][1];
    if (v1 > bv || (v1 == bv && i1 < bi)) { bv = v1; bi = i1; }
    idx_i[blockIdx.x * 64 + r] = bi;
    idx_f[blockIdx.x * 64 + r] = (float)bi;
  }
}

// ---------------------------------------------------------------------------
// Gather zq = emb[idx] (f32 -> bf16 for the final GEMM) + fused commit loss:
// loss = (1+beta)/D * sum((zq - zl)^2). One wave per row.
__global__ __launch_bounds__(256) void gather_loss(
    const int* __restrict__ idx, const float* __restrict__ emb,
    const float* __restrict__ zl, __bf16* __restrict__ zq_bf,
    float* __restrict__ loss) {
  int lane = threadIdx.x & 31, wave = threadIdx.x >> 5;
  int row = blockIdx.x * 8 + wave;
  int code = idx[row];
  const float* ep = emb + (size_t)code * LAT_D + lane * 8;
  const float* lp = zl + (size_t)row * LAT_D + lane * 8;
  __bf16* qp = zq_bf + (size_t)row * LAT_D + lane * 8;
  float acc = 0.0f;
#pragma unroll
  for (int i = 0; i < 8; ++i) {
    float q = ep[i], l = lp[i];
    float d = q - l;
    acc += d * d;
    qp[i] = (__bf16)q;
  }
#pragma unroll
  for (int m = 16; m >= 1; m >>= 1) acc += __shfl_xor(acc, m, 32);
  if (lane == 0) loss[row] = acc * (1.25f / (float)LAT_D);
}

// ---------------------------------------------------------------------------
// GEMM3: out = zq @ Wp + bp.  Per wave: 16 rows x 128 cols, K=256.
// 8 waves/block: 2 M-tiles x 4 col-chunks. grid = N_ROWS/32.
__global__ __launch_bounds__(256) void g3_gemm(
    const __bf16* __restrict__ zq, const __bf16* __restrict__ WpT,
    const float* __restrict__ bp, float* __restrict__ out) {
  int lane = threadIdx.x & 31, wave = threadIdx.x >> 5;
  int row0 = (blockIdx.x * 2 + (wave >> 2)) * 16;
  int col0 = (wave & 3) * 128;
  v8f acc[8] = {};
  for (int k0 = 0; k0 < LAT_D; k0 += 32) {
    v16bf B[8];
#pragma unroll
    for (int t = 0; t < 8; ++t)
      B[t] = load_b_bf16(WpT, LAT_D, col0 + t * 16, k0, lane);
    v16bf a = load_a_bf16(zq, LAT_D, row0, k0, lane);
#pragma unroll
    for (int t = 0; t < 8; ++t) acc[t] = wmma_bf16(a, B[t], acc[t]);
  }
  int h = lane >> 4, n = lane & 15;
#pragma unroll
  for (int t = 0; t < 8; ++t) {
    int col = col0 + t * 16 + n;
    float bias = bp[col];
#pragma unroll
    for (int j = 0; j < 8; ++j)
      out[(size_t)(row0 + 8 * h + j) * OUT_D + col] = acc[t][j] + bias;
  }
}

// ---------------------------------------------------------------------------
extern "C" void kernel_launch(void* const* d_in, const int* in_sizes, int n_in,
                              void* d_out, int out_size, void* d_ws, size_t ws_size,
                              hipStream_t stream) {
  const float* z   = (const float*)d_in[0];
  const float* Wq  = (const float*)d_in[1];
  const float* bq  = (const float*)d_in[2];
  const float* emb = (const float*)d_in[3];
  const float* Wp  = (const float*)d_in[4];
  const float* bp  = (const float*)d_in[5];

  float* out   = (float*)d_out;                       // [16384 x 512]
  float* idx_f = out + (size_t)N_ROWS * OUT_D;        // [16384] (as float)
  float* loss  = idx_f + N_ROWS;                      // [16384]

  char* ws = (char*)d_ws;
  float*  zl  = (float*)ws;   ws += (size_t)N_ROWS * LAT_D * 4;   // 16 MB
  __bf16* zn  = (__bf16*)ws;  ws += (size_t)N_ROWS * LAT_D * 2;   //  8 MB
  __bf16* en  = (__bf16*)ws;  ws += (size_t)N_CODES * LAT_D * 2;  //  4 MB
  __bf16* zq  = (__bf16*)ws;  ws += (size_t)N_ROWS * LAT_D * 2;   //  8 MB
  __bf16* WqT = (__bf16*)ws;  ws += (size_t)IN_D * LAT_D * 2;     // 256 KB
  __bf16* WpT = (__bf16*)ws;  ws += (size_t)LAT_D * OUT_D * 2;    // 256 KB
  int*    idx = (int*)ws;

  transpose_cvt<<<(IN_D * LAT_D + 255) / 256, 256, 0, stream>>>(Wq, WqT, IN_D, LAT_D);
  transpose_cvt<<<(LAT_D * OUT_D + 255) / 256, 256, 0, stream>>>(Wp, WpT, LAT_D, OUT_D);
  row_l2norm_bf16<<<N_CODES / 8, 256, 0, stream>>>(emb, en);
  g1_gemm<<<N_ROWS / 32, 256, 0, stream>>>(z, WqT, bq, zl);
  row_l2norm_bf16<<<N_ROWS / 8, 256, 0, stream>>>(zl, zn);
  g2_argmax<<<N_ROWS / 64, 256, 0, stream>>>(zn, en, idx, idx_f);
  gather_loss<<<N_ROWS / 8, 256, 0, stream>>>(idx, emb, zl, zq, loss);
  g3_gemm<<<N_ROWS / 32, 256, 0, stream>>>(zq, WpT, bp, out);
}
